// PatchMerger_47528108098026
// MI455X (gfx1250) — matrix operand order repaired and checked by reading, not compiled, not run on hardware
//
#include <hip/hip_runtime.h>
#include <hip/hip_bf16.h>

typedef __attribute__((ext_vector_type(16))) __bf16 v16bf;
typedef __attribute__((ext_vector_type(8)))  float  v8f;

constexpr int Bn = 32;    // batch
constexpr int Nn = 4096;  // sequence
constexpr int Dn = 1024;  // feature
constexpr int Mn = 64;    // queries

__device__ __forceinline__ unsigned short f2bf(float f) {
    unsigned int u = __float_as_uint(f);
    unsigned int r = u + 0x7FFFu + ((u >> 16) & 1u);   // round to nearest even
    return (unsigned short)(r >> 16);
}

struct Frag32B { uint4 a, b; };   // 32 bytes == v16bf

// 16 contiguous bf16 (B operand: one column's K-run)
__device__ __forceinline__ v16bf loadFragB(const unsigned short* p) {
    Frag32B f;
    f.a = *reinterpret_cast<const uint4*>(p);
    f.b = *reinterpret_cast<const uint4*>(p + 8);
    return __builtin_bit_cast(v16bf, f);
}
// A operand: 16 contiguous K elements (one row's K-run)
__device__ __forceinline__ v16bf loadFragA(const unsigned short* p) {
    Frag32B f;
    f.a = *reinterpret_cast<const uint4*>(p);
    f.b = *reinterpret_cast<const uint4*>(p + 8);
    return __builtin_bit_cast(v16bf, f);
}

__device__ __forceinline__ v8f wmma_bf16(v16bf a, v16bf b, v8f c) {
    return __builtin_amdgcn_wmma_f32_16x16x32_bf16(
        false, a, false, b, (short)0, c, false, false);
}

// ---------------------------------------------------------------- LayerNorm
__global__ __launch_bounds__(256) void ln_kernel(
    const float* __restrict__ x, const float* __restrict__ w,
    const float* __restrict__ bia, unsigned short* __restrict__ xn) {
    const int row = blockIdx.x;                 // 0 .. B*N-1
    const int t = threadIdx.x;
    const float* xr = x + (size_t)row * Dn;
    float4 v = reinterpret_cast<const float4*>(xr)[t];

    float s  = v.x + v.y + v.z + v.w;
    float s2 = v.x * v.x + v.y * v.y + v.z * v.z + v.w * v.w;
    #pragma unroll
    for (int off = 16; off; off >>= 1) {
        s  += __shfl_xor(s,  off, 32);
        s2 += __shfl_xor(s2, off, 32);
    }
    __shared__ float rs[8], rs2[8];
    const int wave = t >> 5, lane = t & 31;
    if (lane == 0) { rs[wave] = s; rs2[wave] = s2; }
    __syncthreads();
    float ts = 0.f, ts2 = 0.f;
    #pragma unroll
    for (int i = 0; i < 8; ++i) { ts += rs[i]; ts2 += rs2[i]; }
    const float mu  = ts * (1.0f / Dn);
    const float var = ts2 * (1.0f / Dn) - mu * mu;
    const float rstd = rsqrtf(var + 1e-5f);

    float4 wv = reinterpret_cast<const float4*>(w)[t];
    float4 bv = reinterpret_cast<const float4*>(bia)[t];
    ushort4 o = make_ushort4(
        f2bf((v.x - mu) * rstd * wv.x + bv.x),
        f2bf((v.y - mu) * rstd * wv.y + bv.y),
        f2bf((v.z - mu) * rstd * wv.z + bv.z),
        f2bf((v.w - mu) * rstd * wv.w + bv.w));
    reinterpret_cast<ushort4*>(xn + (size_t)row * Dn)[t] = o;
}

// ---------------------------------------------------------- queries -> bf16
__global__ __launch_bounds__(256) void qconv_kernel(
    const float* __restrict__ q, unsigned short* __restrict__ qbf) {
    const int i = blockIdx.x * 256 + threadIdx.x;       // each handles 4
    float4 f = reinterpret_cast<const float4*>(q)[i];
    ushort4 o = make_ushort4(f2bf(f.x), f2bf(f.y), f2bf(f.z), f2bf(f.w));
    reinterpret_cast<ushort4*>(qbf)[i] = o;
}

// ------------------------------------------- GEMM1: attn = Q * xn^T (bf16)
// grid (N/256, B); block 256 = 8 waves; wave covers 64 x (2x16) of attn.
// K loop unrolled 2x with disjoint ping-pong register sets (no copies).
__global__ __launch_bounds__(256) void gemm1_kernel(
    const unsigned short* __restrict__ qbf,
    const unsigned short* __restrict__ xnbf,
    float* __restrict__ attnf) {
    const int b    = blockIdx.y;
    const int n0   = blockIdx.x * 256;
    const int wave = threadIdx.x >> 5;
    const int lane = threadIdx.x & 31;
    const int arow = lane & 15;
    const int agrp = lane >> 4;                         // K-half select
    const int ncol0 = n0 + wave * 16 + (lane & 15);
    const int ncol1 = ncol0 + 128;
    const unsigned short* xb = xnbf + (size_t)b * Nn * Dn;

    const unsigned short* bp0 = xb + (size_t)ncol0 * Dn + agrp * 16;
    const unsigned short* bp1 = xb + (size_t)ncol1 * Dn + agrp * 16;
    const unsigned short* apb = qbf + (size_t)arow * Dn + agrp * 16;

    v8f acc[8] = {};
    v16bf a0[4], a1[4], b00, b01, b10, b11;

    // prologue: set0 <- kb = 0
    b00 = loadFragB(bp0);
    b01 = loadFragB(bp1);
    #pragma unroll
    for (int mt = 0; mt < 4; ++mt) a0[mt] = loadFragA(apb + (size_t)mt * 16 * Dn);

    for (int kb = 0; kb < Dn; kb += 64) {
        // set1 <- kb + 32 (always in-range: Dn % 64 == 0)
        {
            const int kn = kb + 32;
            __builtin_prefetch(bp0 + kn + 32, 0, 1);
            b10 = loadFragB(bp0 + kn);
            b11 = loadFragB(bp1 + kn);
            #pragma unroll
            for (int mt = 0; mt < 4; ++mt)
                a1[mt] = loadFragA(apb + (size_t)mt * 16 * Dn + kn);
        }
        #pragma unroll
        for (int mt = 0; mt < 4; ++mt) {
            acc[mt]     = wmma_bf16(a0[mt], b00, acc[mt]);
            acc[4 + mt] = wmma_bf16(a0[mt], b01, acc[4 + mt]);
        }
        if (kb + 64 < Dn) {                     // set0 <- kb + 64
            const int kn = kb + 64;
            b00 = loadFragB(bp0 + kn);
            b01 = loadFragB(bp1 + kn);
            #pragma unroll
            for (int mt = 0; mt < 4; ++mt)
                a0[mt] = loadFragA(apb + (size_t)mt * 16 * Dn + kn);
        }
        #pragma unroll
        for (int mt = 0; mt < 4; ++mt) {
            acc[mt]     = wmma_bf16(a1[mt], b10, acc[mt]);
            acc[4 + mt] = wmma_bf16(a1[mt], b11, acc[4 + mt]);
        }
    }

    float* ao = attnf + (size_t)b * Mn * Nn;
    const int mbase = agrp * 8;
    #pragma unroll
    for (int mt = 0; mt < 4; ++mt)
        #pragma unroll
        for (int j = 0; j < 8; ++j) {
            const size_t r = (size_t)(mt * 16 + mbase + j) * Nn;
            ao[r + ncol0] = acc[mt][j];
            ao[r + ncol1] = acc[4 + mt][j];
        }
}

// ------------------------------------------------------------ row softmax
__global__ __launch_bounds__(256) void softmax_kernel(
    const float* __restrict__ attnf, unsigned short* __restrict__ attnb) {
    const int row = blockIdx.x;                 // 0 .. B*M-1
    const int t = threadIdx.x;
    const int wave = t >> 5, lane = t & 31;
    const float* r = attnf + (size_t)row * Nn;

    float v[16];
    float mx = -3.0e38f;
    #pragma unroll
    for (int i = 0; i < 4; ++i) {
        float4 f = reinterpret_cast<const float4*>(r)[t + i * 256];
        v[i * 4 + 0] = f.x; v[i * 4 + 1] = f.y;
        v[i * 4 + 2] = f.z; v[i * 4 + 3] = f.w;
        mx = fmaxf(mx, fmaxf(fmaxf(f.x, f.y), fmaxf(f.z, f.w)));
    }
    #pragma unroll
    for (int off = 16; off; off >>= 1) mx = fmaxf(mx, __shfl_xor(mx, off, 32));
    __shared__ float red[8];
    if (lane == 0) red[wave] = mx;
    __syncthreads();
    float bm = red[0];
    #pragma unroll
    for (int i = 1; i < 8; ++i) bm = fmaxf(bm, red[i]);
    __syncthreads();

    float s = 0.f;
    #pragma unroll
    for (int i = 0; i < 16; ++i) { v[i] = __expf(v[i] - bm); s += v[i]; }
    #pragma unroll
    for (int off = 16; off; off >>= 1) s += __shfl_xor(s, off, 32);
    if (lane == 0) red[wave] = s;
    __syncthreads();
    float ssum = 0.f;
    #pragma unroll
    for (int i = 0; i < 8; ++i) ssum += red[i];
    const float inv = 1.0f / ssum;

    unsigned short* ob = attnb + (size_t)row * Nn;
    #pragma unroll
    for (int i = 0; i < 4; ++i) {
        ushort4 o = make_ushort4(f2bf(v[i * 4 + 0] * inv),
                                 f2bf(v[i * 4 + 1] * inv),
                                 f2bf(v[i * 4 + 2] * inv),
                                 f2bf(v[i * 4 + 3] * inv));
        reinterpret_cast<ushort4*>(ob)[t + i * 256] = o;
    }
}

// -------------------------------------------- GEMM2: out = attn * xn (bf16)
// grid (D/256, B); xn K-slabs ping-pong in LDS; 2x-unrolled K loop binds the
// slabs statically so there is no index math or register rotation.
constexpr int LD2 = 264;                       // padded LDS row stride (bf16)

__device__ __forceinline__ void stage_tile(
    unsigned short* __restrict__ dst, const unsigned short* __restrict__ xb,
    int kb, int d0, int t) {
    #pragma unroll
    for (int rep = 0; rep < 4; ++rep) {
        const int idx = t + rep * 256;          // 0..1023 uint4 slots
        const int rr  = idx >> 5;               // 32 uint4 per 256-col row
        const int cc  = (idx & 31) * 8;
        uint4 d = *reinterpret_cast<const uint4*>(
            xb + (size_t)(kb + rr) * Dn + d0 + cc);
        *reinterpret_cast<uint4*>(&dst[rr * LD2 + cc]) = d;
    }
}

__device__ __forceinline__ v16bf gather_col(
    const unsigned short* __restrict__ tile, int k0, int col) {
    struct { unsigned short e[16]; } bs;
    #pragma unroll
    for (int e = 0; e < 16; ++e) bs.e[e] = tile[(k0 + e) * LD2 + col];
    return __builtin_bit_cast(v16bf, bs);
}

__global__ __launch_bounds__(256) void gemm2_kernel(
    const unsigned short* __restrict__ attnb,
    const unsigned short* __restrict__ xnbf,
    float* __restrict__ out) {
    const int b    = blockIdx.y;
    const int d0   = blockIdx.x * 256;
    const int t    = threadIdx.x;
    const int wave = t >> 5;
    const int lane = t & 31;
    const int arow = lane & 15;
    const int agrp = lane >> 4;
    const int ccol0 = wave * 16 + (lane & 15);          // cols within tile
    const int ccol1 = ccol0 + 128;

    __shared__ unsigned short tiles[2][32 * LD2];       // ping-pong K-slabs
    const unsigned short* xb = xnbf + (size_t)b * Nn * Dn;
    const unsigned short* ab = attnb + (size_t)b * Mn * Nn;
    const unsigned short* apb = ab + (size_t)arow * Nn + agrp * 16;

    v8f acc[8] = {};
    v16bf a0[4], a1[4];

    // prologue: slab0 + A set0 <- kb = 0
    stage_tile(tiles[0], xb, 0, d0, t);
    #pragma unroll
    for (int mt = 0; mt < 4; ++mt) a0[mt] = loadFragA(apb + (size_t)mt * 16 * Nn);
    __syncthreads();

    for (int kb = 0; kb < Nn; kb += 64) {
        // slab1 + A set1 <- kb + 32 (always in-range: Nn % 64 == 0)
        stage_tile(tiles[1], xb, kb + 32, d0, t);
        #pragma unroll
        for (int mt = 0; mt < 4; ++mt)
            a1[mt] = loadFragA(apb + (size_t)mt * 16 * Nn + kb + 32);

        {   // compute on slab0 / set0
            const v16bf c0 = gather_col(tiles[0], agrp * 16, ccol0);
            const v16bf c1 = gather_col(tiles[0], agrp * 16, ccol1);
            #pragma unroll
            for (int mt = 0; mt < 4; ++mt) {
                acc[mt]     = wmma_bf16(a0[mt], c0, acc[mt]);
                acc[4 + mt] = wmma_bf16(a0[mt], c1, acc[4 + mt]);
            }
        }
        __syncthreads();            // slab1 staged; slab0 free for rewrite

        if (kb + 64 < Nn) {         // slab0 + A set0 <- kb + 64
            stage_tile(tiles[0], xb, kb + 64, d0, t);
            #pragma unroll
            for (int mt = 0; mt < 4; ++mt)
                a0[mt] = loadFragA(apb + (size_t)mt * 16 * Nn + kb + 64);
        }
        {   // compute on slab1 / set1
            const v16bf c0 = gather_col(tiles[1], agrp * 16, ccol0);
            const v16bf c1 = gather_col(tiles[1], agrp * 16, ccol1);
            #pragma unroll
            for (int mt = 0; mt < 4; ++mt) {
                acc[mt]     = wmma_bf16(a1[mt], c0, acc[mt]);
                acc[4 + mt] = wmma_bf16(a1[mt], c1, acc[4 + mt]);
            }
        }
        __syncthreads();            // slab0 staged; slab1 free for rewrite
    }

    float* ob = out + (size_t)b * Mn * Dn;
    const int mbase = agrp * 8;
    #pragma unroll
    for (int mt = 0; mt < 4; ++mt)
        #pragma unroll
        for (int j = 0; j < 8; ++j) {
            const size_t r = (size_t)(mt * 16 + mbase + j) * Dn + d0;
            ob[r + ccol0] = acc[mt][j];
            ob[r + ccol1] = acc[4 + mt][j];
        }
}

// ---------------------------------------------------------------- launcher
extern "C" void kernel_launch(void* const* d_in, const int* in_sizes, int n_in,
                              void* d_out, int out_size, void* d_ws, size_t ws_size,
                              hipStream_t stream) {
    const float* x  = (const float*)d_in[0];
    const float* q  = (const float*)d_in[1];
    const float* lw = (const float*)d_in[2];
    const float* lb = (const float*)d_in[3];
    float* out = (float*)d_out;

    char* ws = (char*)d_ws;
    const size_t XN_BYTES = (size_t)Bn * Nn * Dn * sizeof(unsigned short); // 256 MB
    const size_t AF_BYTES = (size_t)Bn * Mn * Nn * sizeof(float);          // 32 MB
    const size_t AB_BYTES = (size_t)Bn * Mn * Nn * sizeof(unsigned short); // 16 MB
    unsigned short* xn    = (unsigned short*)ws;
    float*          attnf = (float*)(ws + XN_BYTES);
    unsigned short* attnb = (unsigned short*)(ws + XN_BYTES + AF_BYTES);
    unsigned short* qbf   = (unsigned short*)(ws + XN_BYTES + AF_BYTES + AB_BYTES);

    ln_kernel<<<Bn * Nn, 256, 0, stream>>>(x, lw, lb, xn);
    qconv_kernel<<<(Mn * Dn) / (256 * 4), 256, 0, stream>>>(q, qbf);
    gemm1_kernel<<<dim3(Nn / 256, Bn), 256, 0, stream>>>(qbf, xn, attnf);
    softmax_kernel<<<Bn * Mn, 256, 0, stream>>>(attnf, attnb);
    gemm2_kernel<<<dim3(Dn / 256, Bn), 256, 0, stream>>>(attnb, xn, out);
}
